// LSTM_71012989272111
// MI455X (gfx1250) — matrix operand order repaired and checked by reading, not compiled, not run on hardware
//
#include <hip/hip_runtime.h>
#include <hip/hip_bf16.h>
#include <stdint.h>

#define T_TOT 512
#define B_SZ  256
#define IN_SZ 128
#define H_SZ  256
#define GH    (4 * H_SZ)    // 1024 gate columns (i,f,o,g)
#define HP    (H_SZ + 8)    // padded LDS row (bank-safe, 16B aligned rows)
#define HBUF  (B_SZ * HP)   // one h buffer, elements

typedef __attribute__((ext_vector_type(16))) __bf16 v16bf;
typedef __attribute__((ext_vector_type(8)))  float  v8f;

union FragAB { v16bf v; uint4 q[2]; };             // 32 B: A or B bf16 fragment
union FragC  { v8f  v; float f[8]; float4 x[2]; }; // 32 B: C/D f32 fragment

__device__ __forceinline__ unsigned short f32_to_bf16(float f) {
    union { float f; uint32_t u; } x; x.f = f;
    uint32_t r = x.u + 0x7fffu + ((x.u >> 16) & 1u);   // round-to-nearest-even
    return (unsigned short)(r >> 16);
}

#define LOG2E_F 1.4426950408889634f

__device__ __forceinline__ float fast_sigmoid(float x) {
    // v_mul + v_exp_f32 + v_add + v_rcp_f32 (TRANS ops co-execute with WMMA)
    return __builtin_amdgcn_rcpf(1.0f + __builtin_amdgcn_exp2f(-LOG2E_F * x));
}

__device__ __forceinline__ float fast_tanh(float x) {
#if __has_builtin(__builtin_amdgcn_tanhf)
    return __builtin_amdgcn_tanhf(x);                  // v_tanh_f32 (CDNA5 TRANS op)
#else
    return 1.0f - 2.0f * __builtin_amdgcn_rcpf(1.0f + __builtin_amdgcn_exp2f((2.0f * LOG2E_F) * x));
#endif
}

// ---------------------------------------------------------------------------
// Kernel 0: one-shot prep. bf16 weight copies, combined bias, zero h/c state.
// ---------------------------------------------------------------------------
__global__ void lstm_prep(const float* __restrict__ W_ih, const float* __restrict__ W_hh,
                          const float* __restrict__ b_ih, const float* __restrict__ b_hh,
                          unsigned short* __restrict__ wih_bf, unsigned short* __restrict__ whh_bf,
                          float* __restrict__ bias, unsigned short* __restrict__ h_ws,
                          float* __restrict__ c_ws) {
    int i = blockIdx.x * blockDim.x + threadIdx.x;
    if (i < GH * H_SZ)  whh_bf[i] = f32_to_bf16(W_hh[i]);
    if (i < GH * IN_SZ) wih_bf[i] = f32_to_bf16(W_ih[i]);
    if (i < GH)         bias[i]   = b_ih[i] + b_hh[i];
    if (i < B_SZ * H_SZ) { h_ws[i] = 0; c_ws[i] = 0.0f; }
}

// ---------------------------------------------------------------------------
// Kernel 1: time-parallel input projection. Z stored in WMMA C-fragment order:
//   Z[(strip*64 + jtile)*256 + lane*8 + v],  strip = row/16, jtile = col/16.
// ---------------------------------------------------------------------------
__global__ __launch_bounds__(256)
void lstm_ingemm(const float* __restrict__ x,              // [rows][IN] f32
                 const unsigned short* __restrict__ wih_bf,// [GH][IN] bf16
                 const float* __restrict__ bias,           // [GH]
                 float* __restrict__ Z)                    // swizzled tiles
{
    __shared__ unsigned short xs[16][IN_SZ + 8];           // 136-elem rows
    const int tid = threadIdx.x;
    const int r0  = blockIdx.x * 16;

    for (int e = tid; e < 16 * IN_SZ; e += 256) {
        int r = e >> 7, c = e & 127;
        xs[r][c] = f32_to_bf16(x[(size_t)(r0 + r) * IN_SZ + c]);
    }
    __syncthreads();

    const int lane = tid & 31, wave = tid >> 5;
    const int l16 = lane & 15, hf = lane >> 4;
    const int off8 = hf * 8, kb16 = hf * 16;
    const int bIdx = l16 * IN_SZ + off8;                   // per-lane W_ih index base
    const unsigned short* xrow = &xs[0][0] + l16 * (IN_SZ + 8) + off8;

    for (int q2 = 0; q2 < 8; ++q2) {
        const int n0 = (wave * 8 + q2) * 16;
        FragC acc;
        const float bv = bias[n0 + l16];
        #pragma unroll
        for (int v = 0; v < 8; ++v) acc.f[v] = bv;

        #pragma unroll
        for (int kk = 0; kk < IN_SZ; kk += 32) {
            FragAB a, b;
            a.q[0] = *(const uint4*)(xrow + kk);
            a.q[1] = *(const uint4*)(xrow + kk + 16);
            const uint4* bp = (const uint4*)(wih_bf + bIdx + n0 * IN_SZ + kk);
            b.q[0] = bp[0]; b.q[1] = bp[2];
            acc.v = __builtin_amdgcn_wmma_f32_16x16x32_bf16(
                false, a.v, false, b.v, (short)0, acc.v, false, false);
        }
        float4* zt = (float4*)(Z + ((size_t)blockIdx.x * 64 + wave * 8 + q2) * 256 + lane * 8);
        zt[0] = acc.x[0];
        zt[1] = acc.x[1];
    }
}

// ---------------------------------------------------------------------------
// Kernel 2: persistent recurrent kernel (single workgroup, 32 waves).
// gates = Z[t] + h @ W_hh^T (WMMA bf16) -> LSTM cell -> h to LDS + output.
// h is DOUBLE-buffered in LDS (2 x 132 KB of the 320 KB WGP budget): the cell
// update writes new-h directly into the write buffer, so only ONE barrier per
// step and no registers parked across it.
// ---------------------------------------------------------------------------
__global__ __launch_bounds__(1024)
void lstm_recur(const float* __restrict__ Z,                // swizzled [Tc*16*64][256]
                const unsigned short* __restrict__ whh_bf,  // [GH][H] bf16
                const int* __restrict__ lengths,            // [B]
                float* __restrict__ out,                    // [T][B][H]
                unsigned short* __restrict__ h_ws,          // [B][H] bf16
                float* __restrict__ c_ws,                   // swizzled [256 tiles][256]
                int t0, int Tc)
{
    __shared__ unsigned short hbuf[2][B_SZ][HP];            // 264 KB double buffer
    __shared__ int lens[B_SZ];

    const int tid = threadIdx.x;
    if (tid < B_SZ) lens[tid] = lengths[tid];
    for (int e = tid; e < B_SZ * H_SZ; e += 1024)
        hbuf[0][e >> 8][e & 255] = h_ws[e];

    const int lane = tid & 31, wave = tid >> 5;
    const int l16 = lane & 15, hf = lane >> 4;
    const int off8 = hf * 8, kb16 = hf * 16;

    // per-lane invariant index bases (element units)
    const int wIdx  = l16 * H_SZ + off8;            // into whh_bf
    const int zIdx  = lane * 8;                     // into Z tile
    const int cIdx  = wave * 2048 + lane * 8;       // into c_ws
    const int oIdx  = off8 * H_SZ + l16;            // into out step-slab
    const unsigned short* hrd = &hbuf[0][0][0] + l16 * HP + off8;   // A-frag read base
    unsigned short*       hwr = &hbuf[0][0][0] + off8 * HP + l16;   // h write base

    __syncthreads();

    const float* zp = Z;
    float*       op = out + (size_t)t0 * B_SZ * H_SZ;

    for (int i = 0; i < Tc; ++i) {
        const int t = t0 + i;
        const int rsel = (i & 1) * HBUF;            // read buffer offset
        const int wsel = HBUF - rsel;               // write buffer offset

        #pragma unroll
        for (int q = 0; q < 8; ++q) {
            const int p  = wave * 8 + q;
            const int bt = p >> 4, ht = p & 15;

            FragC ai, af, ao, ag;
            {   // C-init from fragment-order Z: const offsets fold into loads
                const float* zb = zp + zIdx + ((bt * 64 + ht) << 8);
                const float4* zi = (const float4*)(zb + 0 * 16 * 256);
                const float4* zf = (const float4*)(zb + 1 * 16 * 256);
                const float4* zo = (const float4*)(zb + 2 * 16 * 256);
                const float4* zg = (const float4*)(zb + 3 * 16 * 256);
                ai.x[0] = zi[0]; ai.x[1] = zi[1];
                af.x[0] = zf[0]; af.x[1] = zf[1];
                ao.x[0] = zo[0]; ao.x[1] = zo[1];
                ag.x[0] = zg[0]; ag.x[1] = zg[1];
                if (i + 1 < Tc)
                    __builtin_prefetch(zb + 16 * 64 * 256, 0, 1);
            }

            const unsigned short* hrow = hrd + rsel + bt * (16 * HP);

            #pragma unroll
            for (int kk = 0; kk < H_SZ; kk += 32) {
                FragAB a;
                a.q[0] = *(const uint4*)(hrow + kk);
                a.q[1] = *(const uint4*)(hrow + kk + 16);

                FragAB bi, bf2, bo, bg;
                const unsigned short* wb = whh_bf + wIdx + ht * (16 * H_SZ) + kk;
                const uint4* pi = (const uint4*)(wb + 0 * H_SZ * H_SZ);
                const uint4* pf = (const uint4*)(wb + 1 * H_SZ * H_SZ);
                const uint4* po = (const uint4*)(wb + 2 * H_SZ * H_SZ);
                const uint4* pg = (const uint4*)(wb + 3 * H_SZ * H_SZ);
                bi.q[0]  = pi[0]; bi.q[1]  = pi[2];
                bf2.q[0] = pf[0]; bf2.q[1] = pf[2];
                bo.q[0]  = po[0]; bo.q[1]  = po[2];
                bg.q[0]  = pg[0]; bg.q[1]  = pg[2];

                ai.v = __builtin_amdgcn_wmma_f32_16x16x32_bf16(false, a.v, false, bi.v,  (short)0, ai.v, false, false);
                af.v = __builtin_amdgcn_wmma_f32_16x16x32_bf16(false, a.v, false, bf2.v, (short)0, af.v, false, false);
                ao.v = __builtin_amdgcn_wmma_f32_16x16x32_bf16(false, a.v, false, bo.v,  (short)0, ao.v, false, false);
                ag.v = __builtin_amdgcn_wmma_f32_16x16x32_bf16(false, a.v, false, bg.v,  (short)0, ag.v, false, false);
            }

            // LSTM cell: c' = sig(f)*c + sig(i)*tanh(g); h' = sig(o)*tanh(c')
            FragC cold, cnew;
            float4* cp = (float4*)(c_ws + cIdx + q * 256);
            cold.x[0] = cp[0]; cold.x[1] = cp[1];
            float* ob = op + oIdx + bt * (16 * H_SZ) + ht * 16;
            unsigned short* hw = hwr + wsel + bt * (16 * HP) + ht * 16;
            #pragma unroll
            for (int v = 0; v < 8; ++v) {
                const int b = bt * 16 + off8 + v;
                const float iv = fast_sigmoid(ai.f[v]);
                const float fv = fast_sigmoid(af.f[v]);
                const float ov = fast_sigmoid(ao.f[v]);
                const float gv = fast_tanh(ag.f[v]);
                const float cn = fmaf(fv, cold.f[v], iv * gv);
                cnew.f[v] = cn;
                const float hn = ov * fast_tanh(cn);
                hw[v * HP] = f32_to_bf16(hn);       // straight into write buffer
                // reference masks only the OUTPUT; state keeps evolving
                __builtin_nontemporal_store((t < lens[b]) ? hn : 0.0f, ob + v * H_SZ);
            }
            cp[0] = cnew.x[0]; cp[1] = cnew.x[1];
        }

        __syncthreads();   // flip buffers: new h visible, old h free to overwrite

        zp += 16 * 64 * 256;
        op += B_SZ * H_SZ;
    }

    // final h lives in buffer (Tc & 1)
    const unsigned short* hfin = &hbuf[0][0][0] + (Tc & 1) * HBUF;
    for (int e = tid; e < B_SZ * H_SZ; e += 1024)
        h_ws[e] = hfin[(e >> 8) * HP + (e & 255)];
}

// ---------------------------------------------------------------------------
extern "C" void kernel_launch(void* const* d_in, const int* in_sizes, int n_in,
                              void* d_out, int out_size, void* d_ws, size_t ws_size,
                              hipStream_t stream) {
    const float* seqs    = (const float*)d_in[0];   // [T,B,IN]
    const int*   lengths = (const int*)  d_in[1];   // [B]
    const float* W_ih    = (const float*)d_in[2];   // [4H,IN]
    const float* W_hh    = (const float*)d_in[3];   // [4H,H]
    const float* b_ih    = (const float*)d_in[4];   // [4H]
    const float* b_hh    = (const float*)d_in[5];   // [4H]
    float* out = (float*)d_out;                     // [T,B,H]

    char* ws = (char*)d_ws;
    size_t off = 0;
    unsigned short* whh_bf = (unsigned short*)(ws + off); off += (size_t)GH * H_SZ * 2;
    unsigned short* wih_bf = (unsigned short*)(ws + off); off += (size_t)GH * IN_SZ * 2;
    float*          bias   = (float*)(ws + off);          off += (size_t)GH * 4;
    unsigned short* h_ws   = (unsigned short*)(ws + off); off += (size_t)B_SZ * H_SZ * 2;
    float*          c_ws   = (float*)(ws + off);          off += (size_t)B_SZ * H_SZ * 4;
    off = (off + 255) & ~(size_t)255;
    float* Zc = (float*)(ws + off);
    const size_t zavail = (ws_size > off) ? (ws_size - off) : 0;

    int Tc = 1;   // chunk of timesteps whose pre-activations fit in workspace
    for (int cand = T_TOT; cand >= 1; cand >>= 1) {
        if ((size_t)cand * B_SZ * GH * 4 <= zavail) { Tc = cand; break; }
    }

    lstm_prep<<<(GH * H_SZ + 255) / 256, 256, 0, stream>>>(
        W_ih, W_hh, b_ih, b_hh, wih_bf, whh_bf, bias, h_ws, c_ws);

    for (int t0 = 0; t0 < T_TOT; t0 += Tc) {
        lstm_ingemm<<<Tc * 16, 256, 0, stream>>>(
            seqs + (size_t)t0 * B_SZ * IN_SZ, wih_bf, bias, Zc);
        lstm_recur<<<1, 1024, 0, stream>>>(
            Zc, whh_bf, lengths, out, h_ws, c_ws, t0, Tc);
    }
}
